// ArcticAttention_2113123910129
// MI455X (gfx1250) — compile-verified, matrix operand order checked by reading, not run on hardware
//
#include <hip/hip_runtime.h>

// ---------------- problem constants (from reference) ----------------
#define B_    2
#define S_    2048
#define HID_  4096
#define H_    32
#define HKV_  8
#define DH_   128
#define G_    4
#define WIN_  1024
#define SCALE_ 0.088388347648318447f   // 1/sqrt(128)

typedef __bf16 bf16_t;
typedef __attribute__((ext_vector_type(16))) __bf16 v16bf;
typedef __attribute__((ext_vector_type(8)))  float  v8f;

// TDM descriptor vector types (6-arg builtin signature on this toolchain)
typedef unsigned int u32v4 __attribute__((ext_vector_type(4)));
typedef int          i32v8 __attribute__((ext_vector_type(8)));
typedef int          i32v4 __attribute__((ext_vector_type(4)));

// =====================================================================
// Tiled WMMA GEMM:  C[M,N] = A[M,K] @ W[N,K]^T
//   AMODE: 0 = A is f32 (convert to bf16 on the fly)
//          1 = A is bf16 (A tile staged via Tensor Data Mover DMA)
//   OMODE: 0 = f32 out row-major [M,N]
//          1 = bf16 out row-major [M,N]
//          2 = bf16 out, V-transposed: col n -> (kvh = n/128, dh = n%128),
//              stored at [((b*HKV+kvh)*DH+dh)*S + s], row m = b*S+s
// Block: 256 threads (8 waves). Tile 128x128, BK=32.
// Wave w: rows (w%4)*32..+31, cols (w/4)*64..+63  -> 2x4 WMMA 16x16 tiles.
// =====================================================================
template<int AMODE, int OMODE>
__global__ __launch_bounds__(256)
void wmma_gemm_nt(const void* __restrict__ Aptr,
                  const float* __restrict__ W,
                  void* __restrict__ Cptr,
                  int M, int N, int K)
{
    __shared__ bf16_t sA[128 * 32];
    __shared__ bf16_t sB[128 * 32];

    const int t     = threadIdx.x;
    const int lane  = t & 31;
    const int wid   = t >> 5;
    const int wRow  = (wid & 3) * 32;
    const int wCol  = (wid >> 2) * 64;
    const int l16   = lane & 15;
    const int hh    = lane >> 4;          // 0/1 : which 16-lane half

    const int tileN = blockIdx.x * 128;
    const int tileM = blockIdx.y * 128;

    // cooperative-load mapping: each thread moves 16 contiguous K elems
    const int ldRow = t >> 1;             // 0..127
    const int ldSeg = (t & 1) << 4;       // 0 or 16

    v8f acc[2][4] = {};

    for (int k0 = 0; k0 < K; k0 += 32) {
        // ---- stage A tile ----
        if (AMODE == 1) {
            // Tensor Data Mover: DMA the 128x32 bf16 tile straight into LDS.
            // D# per CDNA5 ISA: 2-D tensor, data_size=2B, tile 32(x) x 128(y).
            if (wid == 0) {
                const bf16_t* Abf = (const bf16_t*)Aptr;
                const unsigned long long ga =
                    (unsigned long long)(uintptr_t)(Abf + (size_t)tileM * K + k0);
                const unsigned int ldsa = (unsigned int)(uintptr_t)(&sA[0]);
                u32v4 g0;
                g0.x = 1u;                                  // count=1, user mode
                g0.y = ldsa;                                // lds_addr
                g0.z = (unsigned int)ga;                    // global_addr[31:0]
                g0.w = (unsigned int)((ga >> 32) & 0x1FFFFFFu) | (2u << 30); // ga[56:32] | type=2
                i32v8 g1;
                g1[0] = 0x10000;                            // data_size=1 (2 bytes), wg_mask=0
                g1[1] = (int)((unsigned)(K & 0xFFFF) << 16);            // tensor_dim0[15:0]
                g1[2] = (int)(((unsigned)K >> 16) |
                              ((unsigned)(M & 0xFFFF) << 16));          // dim0[31:16] | dim1[15:0]
                g1[3] = (int)(((unsigned)M >> 16) | (32u << 16));       // dim1[31:16] | tile_dim0=32
                g1[4] = 128;                                // tile_dim1=128, tile_dim2=0
                g1[5] = K;                                  // tensor_dim0_stride[31:0]
                g1[6] = 0;                                  // stride hi bits, dim1_stride lo
                g1[7] = 0;
                const i32v4 z4 = {0, 0, 0, 0};
                const i32v8 z8 = {0, 0, 0, 0, 0, 0, 0, 0};
                __builtin_amdgcn_tensor_load_to_lds(g0, g1, z4, z4, z8, 0);
            }
        } else {
            const size_t base = (size_t)(tileM + ldRow) * K + k0 + ldSeg;
            alignas(16) bf16_t tmp[16];
            const float* A = (const float*)Aptr;
            if (k0 + 32 < K) __builtin_prefetch(A + base + 32, 0, 1);
            #pragma unroll
            for (int i = 0; i < 16; ++i) tmp[i] = (bf16_t)A[base + i];
            uint4* dst = (uint4*)&sA[ldRow * 32 + ldSeg];
            dst[0] = *(const uint4*)&tmp[0];
            dst[1] = *(const uint4*)&tmp[8];
        }
        // ---- stage W tile (always f32 -> bf16) ----
        {
            const size_t base = (size_t)(tileN + ldRow) * K + k0 + ldSeg;
            alignas(16) bf16_t tmp[16];
            if (k0 + 32 < K) __builtin_prefetch(W + base + 32, 0, 1);
            #pragma unroll
            for (int i = 0; i < 16; ++i) tmp[i] = (bf16_t)W[base + i];
            uint4* dst = (uint4*)&sB[ldRow * 32 + ldSeg];
            dst[0] = *(const uint4*)&tmp[0];
            dst[1] = *(const uint4*)&tmp[8];
        }
        if (AMODE == 1 && wid == 0) {
            __builtin_amdgcn_s_wait_tensorcnt(0);   // TDM tile landed in LDS
        }
        __syncthreads();

        v16bf af[2], bfr[4];
        #pragma unroll
        for (int m = 0; m < 2; ++m)
            af[m] = *(const v16bf*)&sA[(wRow + m * 16 + l16) * 32 + hh * 16];
        #pragma unroll
        for (int n = 0; n < 4; ++n)
            bfr[n] = *(const v16bf*)&sB[(wCol + n * 16 + l16) * 32 + hh * 16];

        #pragma unroll
        for (int m = 0; m < 2; ++m)
            #pragma unroll
            for (int n = 0; n < 4; ++n)
                acc[m][n] = __builtin_amdgcn_wmma_f32_16x16x32_bf16(
                    false, af[m], false, bfr[n], (short)0, acc[m][n], false, false);

        __syncthreads();
    }

    // ---- epilogue ----
    #pragma unroll
    for (int m = 0; m < 2; ++m) {
        #pragma unroll
        for (int n = 0; n < 4; ++n) {
            #pragma unroll
            for (int r = 0; r < 8; ++r) {
                const int grow = tileM + wRow + m * 16 + r + hh * 8;
                const int gcol = tileN + wCol + n * 16 + l16;
                const float val = acc[m][n][r];
                if (OMODE == 0) {
                    ((float*)Cptr)[(size_t)grow * N + gcol] = val;
                } else if (OMODE == 1) {
                    ((bf16_t*)Cptr)[(size_t)grow * N + gcol] = (bf16_t)val;
                } else {
                    const int bb  = grow >> 11;        // /S_
                    const int ss  = grow & (S_ - 1);
                    const int kvh = gcol >> 7;         // /DH_
                    const int dh  = gcol & (DH_ - 1);
                    ((bf16_t*)Cptr)[(((size_t)bb * HKV_ + kvh) * DH_ + dh) * S_ + ss]
                        = (bf16_t)val;
                }
            }
        }
    }
}

// =====================================================================
// RoPE (in place on bf16 buffer laid out [(b*S+s)*nh + h]*DH)
// =====================================================================
__global__ __launch_bounds__(256)
void rope_kernel(void* __restrict__ bufp, int nh, long long total)
{
    const long long idx = (long long)blockIdx.x * 256 + threadIdx.x;
    if (idx >= total) return;
    bf16_t* buf = (bf16_t*)bufp;
    const int d   = (int)(idx & 63);
    long long rem = idx >> 6;
    const int h   = (int)(rem % nh);
    const long long bs = rem / nh;            // b*S + s
    const int pos = (int)(bs & (S_ - 1));
    bf16_t* p = buf + ((size_t)bs * nh + h) * DH_;
    const float inv_freq = __powf(10000.0f, -(float)d * (1.0f / 64.0f));
    const float ang = (float)pos * inv_freq;
    float sn, cs;
    __sincosf(ang, &sn, &cs);
    const float x1 = (float)p[d];
    const float x2 = (float)p[d + 64];
    p[d]      = (bf16_t)(x1 * cs - x2 * sn);
    p[d + 64] = (bf16_t)(x2 * cs + x1 * sn);
}

// =====================================================================
// Flash attention, sliding-window causal, GQA.
// Block = 64 threads (2 waves). Each wave owns 16 Q rows of one (b, qh).
// Grid: (S/32, H, B).  Q:[B,S,H,DH] bf16, K:[B,S,HKV,DH] bf16,
// Vt:[B,HKV,DH,S] bf16, ctx:[B,S,H*DH] bf16.
// =====================================================================
__global__ __launch_bounds__(64)
void flash_attn_kernel(const void* __restrict__ Qp, const void* __restrict__ Kp,
                       const void* __restrict__ Vp, void* __restrict__ Cp)
{
    const bf16_t* Q  = (const bf16_t*)Qp;
    const bf16_t* Kb = (const bf16_t*)Kp;
    const bf16_t* Vt = (const bf16_t*)Vp;
    bf16_t*       ctx = (bf16_t*)Cp;

    __shared__ bf16_t pLds[2][16 * 32];

    const int wave = threadIdx.x >> 5;
    const int lane = threadIdx.x & 31;
    const int l16  = lane & 15;
    const int hh   = lane >> 4;

    const int qh  = blockIdx.y;
    const int b   = blockIdx.z;
    const int kvh = qh >> 2;                       // /G_
    const int qr0 = blockIdx.x * 32 + wave * 16;

    // Q fragments (16 rows x DH, 4 chunks of K=32)
    v16bf aQ[4];
    {
        const bf16_t* qp = Q + ((size_t)(b * S_ + qr0 + l16) * H_ + qh) * DH_;
        #pragma unroll
        for (int c = 0; c < 4; ++c)
            aQ[c] = *(const v16bf*)(qp + c * 32 + hh * 16);
    }

    v8f accO[8] = {};
    float mrun[8], lrun[8];
    #pragma unroll
    for (int r = 0; r < 8; ++r) { mrun[r] = -1e30f; lrun[r] = 0.0f; }

    const int lo = qr0 - (WIN_ - 1);
    const int kbStart = (lo > 0 ? lo : 0) >> 5;
    const int kbEnd   = (qr0 + 15) >> 5;

    for (int kb = kbStart; kb <= kbEnd; ++kb) {
        const int j0 = kb * 32;

        // K fragments: two 16-key column groups, 4 dh-chunks each
        v16bf bK[2][4];
        #pragma unroll
        for (int n = 0; n < 2; ++n) {
            const bf16_t* kp = Kb + ((size_t)(b * S_ + j0 + n * 16 + l16) * HKV_ + kvh) * DH_;
            #pragma unroll
            for (int c = 0; c < 4; ++c)
                bK[n][c] = *(const v16bf*)(kp + c * 32 + hh * 16);
        }

        // scores S = Q @ K^T  (16 x 32)
        v8f accS[2] = {};
        #pragma unroll
        for (int c = 0; c < 4; ++c)
            #pragma unroll
            for (int n = 0; n < 2; ++n)
                accS[n] = __builtin_amdgcn_wmma_f32_16x16x32_bf16(
                    false, aQ[c], false, bK[n][c], (short)0, accS[n], false, false);

        // online softmax per row
        #pragma unroll
        for (int r = 0; r < 8; ++r) {
            const int row = qr0 + r + hh * 8;
            float rm = -1e30f;
            #pragma unroll
            for (int n = 0; n < 2; ++n) {
                const int col = j0 + n * 16 + l16;
                float s = accS[n][r] * SCALE_;
                if (col > row || (row - col) >= WIN_) s = -1e30f;
                accS[n][r] = s;
                rm = fmaxf(rm, s);
            }
            #pragma unroll
            for (int off = 8; off >= 1; off >>= 1)
                rm = fmaxf(rm, __shfl_xor(rm, off, 16));

            const float mn = fmaxf(mrun[r], rm);
            const float alpha = __expf(mrun[r] - mn);
            mrun[r] = mn;

            float psum = 0.0f;
            #pragma unroll
            for (int n = 0; n < 2; ++n) {
                const float p = __expf(accS[n][r] - mn);
                psum += p;
                pLds[wave][(r + hh * 8) * 32 + n * 16 + l16] = (bf16_t)p;
            }
            #pragma unroll
            for (int off = 8; off >= 1; off >>= 1)
                psum += __shfl_xor(psum, off, 16);
            lrun[r] = lrun[r] * alpha + psum;

            #pragma unroll
            for (int dn = 0; dn < 8; ++dn) accO[dn][r] *= alpha;
        }

        // make LDS-written P visible before the transposed fragment read
        asm volatile("s_wait_dscnt 0x0" ::: "memory");
        const v16bf aP = *(const v16bf*)&pLds[wave][l16 * 32 + hh * 16];

        // O += P @ V   (V pre-transposed: contiguous along keys)
        #pragma unroll
        for (int dn = 0; dn < 8; ++dn) {
            const bf16_t* vp = Vt + (((size_t)b * HKV_ + kvh) * DH_ + dn * 16 + l16) * S_
                               + j0 + hh * 16;
            const v16bf bV = *(const v16bf*)vp;
            accO[dn] = __builtin_amdgcn_wmma_f32_16x16x32_bf16(
                false, aP, false, bV, (short)0, accO[dn], false, false);
        }
    }

    // epilogue: normalize and store ctx bf16 [B,S,HID]
    #pragma unroll
    for (int dn = 0; dn < 8; ++dn) {
        #pragma unroll
        for (int r = 0; r < 8; ++r) {
            const int row = qr0 + r + hh * 8;
            const float v = accO[dn][r] / lrun[r];
            ctx[(size_t)(b * S_ + row) * HID_ + qh * DH_ + dn * 16 + l16] = (bf16_t)v;
        }
    }
}

// =====================================================================
// host launcher
// =====================================================================
extern "C" void kernel_launch(void* const* d_in, const int* in_sizes, int n_in,
                              void* d_out, int out_size, void* d_ws, size_t ws_size,
                              hipStream_t stream)
{
    const float* hidden = (const float*)d_in[0];
    const float* wq     = (const float*)d_in[1];
    const float* wk     = (const float*)d_in[2];
    const float* wv     = (const float*)d_in[3];
    const float* wo     = (const float*)d_in[4];
    // d_in[5] attention_mask and d_in[6] position_ids are reproduced analytically.

    const int M  = B_ * S_;              // 4096
    const size_t nQ  = (size_t)M * H_   * DH_;   // 16,777,216
    const size_t nKV = (size_t)M * HKV_ * DH_;   //  4,194,304

    char* ws = (char*)d_ws;
    void* qbuf = (void*)(ws);
    void* kbuf = (void*)(ws + nQ * 2);
    void* vbuf = (void*)(ws + (nQ + nKV) * 2);
    void* cbuf = (void*)(ws + (nQ + 2 * nKV) * 2);

    dim3 blk(256);

    // Q / K projections (bf16 out, row-major); V projection (bf16, key-transposed)
    wmma_gemm_nt<0, 1><<<dim3(HID_ / 128, M / 128), blk, 0, stream>>>(hidden, wq, qbuf, M, HID_, HID_);
    wmma_gemm_nt<0, 1><<<dim3((HKV_ * DH_) / 128, M / 128), blk, 0, stream>>>(hidden, wk, kbuf, M, HKV_ * DH_, HID_);
    wmma_gemm_nt<0, 2><<<dim3((HKV_ * DH_) / 128, M / 128), blk, 0, stream>>>(hidden, wv, vbuf, M, HKV_ * DH_, HID_);

    // RoPE on Q and K
    {
        const long long totQ = (long long)M * H_ * 64;
        const long long totK = (long long)M * HKV_ * 64;
        rope_kernel<<<dim3((unsigned)((totQ + 255) / 256)), blk, 0, stream>>>(qbuf, H_, totQ);
        rope_kernel<<<dim3((unsigned)((totK + 255) / 256)), blk, 0, stream>>>(kbuf, HKV_, totK);
    }

    // sliding-window flash attention
    flash_attn_kernel<<<dim3(S_ / 32, H_, B_), dim3(64), 0, stream>>>(qbuf, kbuf, vbuf, cbuf);

    // O projection (A tile staged by Tensor Data Mover) -> f32 output
    wmma_gemm_nt<1, 0><<<dim3(HID_ / 128, M / 128), blk, 0, stream>>>(cbuf, wo, d_out, M, HID_, HID_);
}
